// stft_15040975471244
// MI455X (gfx1250) — matrix operand order repaired and checked by reading, not compiled, not run on hardware
//
#include <hip/hip_runtime.h>

typedef __attribute__((ext_vector_type(16))) _Float16 v16h;
typedef __attribute__((ext_vector_type(8)))  float    v8f;

#define NFFT   1024
#define HOP    512
#define NBATCH 32
#define NFREQ  513          // nfft/2 + 1
#define NT     511          // n_frames
#define FTILES 33           // ceil(513/16)
#define KCH    32           // 1024 / 32
#define TSAMP  262144

// ---------------------------------------------------------------------------
// Pre-pass: pack the DFT kernels (f32, row-major [f][n]) into WMMA A-fragment
// layout (f16), fragment-major: [ftile][kchunk][lane][16 halves].
// A 16x32 order used here (contiguous k): lane<16 -> M=lane, K offs {0..15};
//                        lane>=16 -> M=lane-16, K offs {16..31}.
// Half j maps to K offset: kbase + j, kbase = kc*32 + (lane>=16?16:0)
// ---------------------------------------------------------------------------
__global__ __launch_bounds__(256) void stft_pack_kernels(
    const float* __restrict__ realk, const float* __restrict__ imagk,
    _Float16* __restrict__ packRe, _Float16* __restrict__ packIm)
{
    int gid = blockIdx.x * blockDim.x + threadIdx.x;   // (ftile,kc,lane)
    if (gid >= FTILES * KCH * 32) return;
    int lane = gid & 31;
    int kc   = (gid >> 5) & 31;
    int ft   = gid >> 10;

    int f     = ft * 16 + (lane & 15);
    int kbase = kc * 32 + ((lane >> 4) << 4);

    v16h vr = {}, vi = {};
    if (f < NFREQ) {
        const float* rr = realk + (size_t)f * NFFT;
        const float* ii = imagk + (size_t)f * NFFT;
#pragma unroll
        for (int j = 0; j < 16; ++j) {
            int koff = kbase + j;
            vr[j] = (_Float16)rr[koff];
            vi[j] = (_Float16)ii[koff];
        }
    }
    size_t o = (size_t)gid * 16;
    *(v16h*)(packRe + o) = vr;
    *(v16h*)(packIm + o) = vi;
}

// ---------------------------------------------------------------------------
// Main GEMM: C[b,f,t] = sum_n sample[b, 512*t + n] * K[f,n]  (re & im)
// Block = 128 threads = 4 waves. Wave w owns 16f x 32t (two 16x16 subtiles).
// Grid = (33 ftiles, 4 t-groups of 128, 32 batches).
// B fragment (32x16 f16): lane n<16 -> col t0+n, K=kc*32+0..15 (contiguous!)
//                         lane n+16 -> col t0+n, K=kc*32+16..31 (contiguous!)
// -> four float4 loads of K-contiguous samples + packed f32->f16 converts.
// Out-of-range t columns: address is CLAMPED to t=NT-1 (valid memory); the
// resulting garbage column is never stored, so the K loop is branch-free and
// EXEC stays full for every WMMA.
// ---------------------------------------------------------------------------
__global__ __launch_bounds__(128) void stft_wmma_kernel(
    const float*    __restrict__ sample,
    const _Float16* __restrict__ packRe,
    const _Float16* __restrict__ packIm,
    float* __restrict__ magn, float* __restrict__ phase, float* __restrict__ ac)
{
    const int lane = threadIdx.x & 31;
    const int wave = threadIdx.x >> 5;
    const int ft   = blockIdx.x;                 // freq tile
    const int b    = blockIdx.z;                 // batch
    const int f0   = ft * 16;
    const int t0w  = blockIdx.y * 128 + wave * 32;   // wave's 32-wide t range

    // Per-lane B addressing (constant over the K loop except the kc*32 term)
    const int nCol  = lane & 15;
    const int kHalf = (lane >> 4) << 4;          // 0 or 16
    const int tB0   = t0w + nCol;                // subtile 0 column
    const int tB1   = t0w + 16 + nCol;           // subtile 1 column
    const int tB0c  = tB0 < NT ? tB0 : (NT - 1); // clamp: garbage col discarded
    const int tB1c  = tB1 < NT ? tB1 : (NT - 1);
    const float* base0 = sample + (size_t)b * TSAMP + (size_t)tB0c * HOP + kHalf;
    const float* base1 = sample + (size_t)b * TSAMP + (size_t)tB1c * HOP + kHalf;

    const v16h* aRe = (const v16h*)packRe + ((size_t)ft * KCH) * 32 + lane;
    const v16h* aIm = (const v16h*)packIm + ((size_t)ft * KCH) * 32 + lane;

    v8f cre0 = {}, cre1 = {}, cim0 = {}, cim1 = {};

#pragma unroll 2
    for (int kc = 0; kc < KCH; ++kc) {
        // A fragments: 32B per lane -> 2x global_load_b128 each
        v16h are = aRe[(size_t)kc * 32];
        v16h aim = aIm[(size_t)kc * 32];

        // B fragments: 16 K-contiguous f32 samples per lane -> f16
        v16h b0, b1;
        {
            const float4* p = (const float4*)(base0 + kc * 32);
#pragma unroll
            for (int i = 0; i < 4; ++i) {
                float4 q = p[i];
                b0[4*i+0] = (_Float16)q.x; b0[4*i+1] = (_Float16)q.y;
                b0[4*i+2] = (_Float16)q.z; b0[4*i+3] = (_Float16)q.w;
            }
        }
        {
            const float4* p = (const float4*)(base1 + kc * 32);
#pragma unroll
            for (int i = 0; i < 4; ++i) {
                float4 q = p[i];
                b1[4*i+0] = (_Float16)q.x; b1[4*i+1] = (_Float16)q.y;
                b1[4*i+2] = (_Float16)q.z; b1[4*i+3] = (_Float16)q.w;
            }
        }

        // 4 x v_wmma_f32_16x16x32_f16 (EXEC full: loop is branch-free)
        cre0 = __builtin_amdgcn_wmma_f32_16x16x32_f16(false, are, false, b0,
                                                      (short)0, cre0, false, false);
        cim0 = __builtin_amdgcn_wmma_f32_16x16x32_f16(false, aim, false, b0,
                                                      (short)0, cim0, false, false);
        cre1 = __builtin_amdgcn_wmma_f32_16x16x32_f16(false, are, false, b1,
                                                      (short)0, cre1, false, false);
        cim1 = __builtin_amdgcn_wmma_f32_16x16x32_f16(false, aim, false, b1,
                                                      (short)0, cim1, false, false);
    }

    // C/D layout: VGPR r, lanes 0-15 -> (M=r, N=lane); lanes 16-31 -> (M=r+8)
    const int mhi = (lane >> 4) << 3;
#pragma unroll
    for (int s = 0; s < 2; ++s) {
        const int t = t0w + s * 16 + nCol;
        if (t >= NT) continue;
#pragma unroll
        for (int r = 0; r < 8; ++r) {
            const int f = f0 + r + mhi;
            const float re = s ? cre1[r] : cre0[r];
            const float im = s ? cim1[r] : cim0[r];
            if (f == 0) {
                ac[(size_t)b * NT + t] = re;                 // DC row -> ac
            } else if (f < NFREQ) {
                const size_t o = ((size_t)b * (NFREQ - 1) + (f - 1)) * NT + t;
                magn[o]  = re;
                phase[o] = -im;                              // phase = -imag
            }
        }
    }
}

extern "C" void kernel_launch(void* const* d_in, const int* in_sizes, int n_in,
                              void* d_out, int out_size, void* d_ws, size_t ws_size,
                              hipStream_t stream) {
    const float* sample = (const float*)d_in[0];
    const float* realk  = (const float*)d_in[1];
    const float* imagk  = (const float*)d_in[2];
    // d_in[3] = hop_length scalar (compile-time constant HOP here)

    const size_t nMagn = (size_t)NBATCH * (NFREQ - 1) * NT;  // 32*512*511
    float* magn  = (float*)d_out;
    float* phase = magn + nMagn;
    float* ac    = phase + nMagn;

    const size_t packElems = (size_t)FTILES * KCH * 32 * 16; // halves per matrix
    _Float16* packRe = (_Float16*)d_ws;
    _Float16* packIm = packRe + packElems;

    // 1) pack DFT kernels into WMMA A-fragment layout (f16) in workspace
    const int packThreads = FTILES * KCH * 32;               // 33792
    stft_pack_kernels<<<(packThreads + 255) / 256, 256, 0, stream>>>(
        realk, imagk, packRe, packIm);

    // 2) WMMA STFT GEMM
    dim3 grid(FTILES, 4, NBATCH);
    stft_wmma_kernel<<<grid, 128, 0, stream>>>(sample, packRe, packIm,
                                               magn, phase, ac);
}
